// MultiHeadSelfAttention_75050258530495
// MI455X (gfx1250) — matrix operand order repaired and checked by reading, not compiled, not run on hardware
//
#include <hip/hip_runtime.h>
#include <hip/hip_bf16.h>

typedef __attribute__((ext_vector_type(16))) _Float16 v16h;
typedef __attribute__((ext_vector_type(8)))  _Float16 v8h;
typedef __attribute__((ext_vector_type(8)))  float    v8f;
typedef __attribute__((ext_vector_type(4)))  unsigned int u32x4;
typedef __attribute__((ext_vector_type(8)))  int i32x8;
typedef __attribute__((ext_vector_type(4)))  int i32x4;

#define WMMA_F32_F16(a,b,c) \
  __builtin_amdgcn_wmma_f32_16x16x32_f16(false,(a),false,(b),(short)0,(c),false,false)

#if __has_builtin(__builtin_amdgcn_tensor_load_to_lds) && \
    __has_builtin(__builtin_amdgcn_s_wait_tensorcnt)
#define USE_TDM 1
#else
#define USE_TDM 0
#endif

#if USE_TDM
// ---------------------------------------------------------------------------
// TDM 2D f16 tile load global->LDS (CDNA5 ISA).
// Pads 4 DWORDs (8 halves) after every 32 DWORDs (64 halves): LDS stride 72.
// tileD0 elems per row (must stay 64 here to match the pad config), tileD1
// rows, rowStride = tensor dim0 stride in elements.
// ---------------------------------------------------------------------------
__device__ __forceinline__ void tdm_load_2d_f16(const _Float16* gsrc,
                                                unsigned int ldsAddr,
                                                unsigned int tileD0,
                                                unsigned int tileD1,
                                                unsigned int rowStride) {
  const unsigned long long ga = (unsigned long long)(uintptr_t)gsrc;
  u32x4 g0;
  g0[0] = 1u;                                   // count=1 (valid user D#)
  g0[1] = ldsAddr;                              // LDS byte address
  g0[2] = (unsigned int)ga;                     // global_addr[31:0]
  g0[3] = (unsigned int)((ga >> 32) & 0x01FFFFFFu) | (2u << 30);  // addr[56:32], type=2
  i32x8 g1;
  g1[0] = (1 << 16)      // data_size = 1 -> 2 bytes
        | (1 << 20)      // pad_enable
        | (4 << 22)      // pad_interval: 32 DWORDs
        | (3 << 25);     // pad_amount:   4 DWORDs
  g1[1] = (int)((tileD0 & 0xFFFFu) << 16);                          // tensor_dim0 lo16
  g1[2] = (int)(((tileD0 >> 16) & 0xFFFFu) | ((tileD1 & 0xFFFFu) << 16)); // d0 hi / d1 lo
  g1[3] = (int)(((tileD1 >> 16) & 0xFFFFu) | ((tileD0 & 0xFFFFu) << 16)); // d1 hi / tile_dim0
  g1[4] = (int)(tileD1 & 0xFFFFu);                                  // tile_dim1 (tile_dim2=0)
  g1[5] = (int)rowStride;                                           // tensor_dim0_stride lo32
  g1[6] = 0;                                                        // stride hi / dim1_stride
  g1[7] = 0;
  i32x4 z4 = {0, 0, 0, 0};
#if __has_include(<hip/amd_detail/amd_gfx1250_TDM.h>)
  i32x8 z8 = {0, 0, 0, 0, 0, 0, 0, 0};
  __builtin_amdgcn_tensor_load_to_lds(g0, g1, z4, z4, z8, 0);   // 6-arg toolchain
#else
  __builtin_amdgcn_tensor_load_to_lds(g0, g1, z4, z4, 0);       // 5-arg (ROCm 7.2)
#endif
}
#endif  // USE_TDM

// ---------------------------------------------------------------------------
// Fragment gathers (per CDNA5 ISA VGPR layouts, wave32)
__device__ __forceinline__ v16h frag_a(const _Float16* base, int stride, int lane) {
  const int r = lane & 15, kh = lane >> 4;
  const _Float16* p = base + r * stride + kh * 8;
  v8h lo = *(const v8h*)(p);
  v8h hi = *(const v8h*)(p + 16);
  v16h o;
#pragma unroll
  for (int i = 0; i < 8; ++i) { o[i] = lo[i]; o[i + 8] = hi[i]; }
  return o;
}

__device__ __forceinline__ v16h frag_b(const _Float16* base, int stride, int lane) {
  const int n = lane & 15, kh = lane >> 4;
  const _Float16* p = base + n * stride + kh * 8;
  v8h lo = *(const v8h*)(p);
  v8h hi = *(const v8h*)(p + 16);
  v16h o;
#pragma unroll
  for (int i = 0; i < 8; ++i) { o[i] = lo[i]; o[i + 8] = hi[i]; }
  return o;
}

__device__ __forceinline__ v8h cvt8(float4 a, float4 b) {
  v8h o;
  o[0] = (_Float16)a.x; o[1] = (_Float16)a.y; o[2] = (_Float16)a.z; o[3] = (_Float16)a.w;
  o[4] = (_Float16)b.x; o[5] = (_Float16)b.y; o[6] = (_Float16)b.z; o[7] = (_Float16)b.w;
  return o;
}

// ---------------------------------------------------------------------------
// Kernel 1: qkv = x @ Wqkv^T + bqkv ; scatter to Q(*0.125)/K/V f16 [B,H,N,64]
// ---------------------------------------------------------------------------
__global__ __launch_bounds__(256) void qkv_gemm(
    const float* __restrict__ x, const float* __restrict__ W,
    const float* __restrict__ bias,
    _Float16* __restrict__ Q, _Float16* __restrict__ K, _Float16* __restrict__ V) {
  __shared__ alignas(16) _Float16 As[128 * 72];
  __shared__ alignas(16) _Float16 Bs[128 * 72];

  const int tid = threadIdx.x, lane = tid & 31, wave = tid >> 5;
  const int wm = wave & 3, wn = wave >> 2;
  const int tm = blockIdx.y * 128, tn = blockIdx.x * 128;

  v8f acc[2][4];
#pragma unroll
  for (int i = 0; i < 2; ++i)
#pragma unroll
    for (int j = 0; j < 4; ++j) acc[i][j] = v8f{};

  const int lrow = tid >> 1, lcol = (tid & 1) * 32;
  const float* xa = x + (size_t)(tm + lrow) * 1024 + lcol;
  const float* wb = W + (size_t)(tn + lrow) * 1024 + lcol;
  _Float16* asp = As + lrow * 72 + lcol;
  _Float16* bsp = Bs + lrow * 72 + lcol;

  for (int k0 = 0; k0 < 1024; k0 += 64) {
    float4 fa[8], fb[8];
#pragma unroll
    for (int q = 0; q < 8; ++q) fa[q] = ((const float4*)(xa + k0))[q];
#pragma unroll
    for (int q = 0; q < 8; ++q) fb[q] = ((const float4*)(wb + k0))[q];
#pragma unroll
    for (int q = 0; q < 4; ++q) {
      *(v8h*)(asp + q * 8) = cvt8(fa[2 * q], fa[2 * q + 1]);
      *(v8h*)(bsp + q * 8) = cvt8(fb[2 * q], fb[2 * q + 1]);
    }
    __syncthreads();

#pragma unroll
    for (int t = 0; t < 2; ++t) {
      v16h af0 = frag_a(As + (wm * 32 + 0)  * 72 + t * 32, 72, lane);
      v16h af1 = frag_a(As + (wm * 32 + 16) * 72 + t * 32, 72, lane);
#pragma unroll
      for (int j = 0; j < 4; ++j) {
        v16h bf = frag_b(Bs + (wn * 64 + j * 16) * 72 + t * 32, 72, lane);
        acc[0][j] = WMMA_F32_F16(af0, bf, acc[0][j]);
        acc[1][j] = WMMA_F32_F16(af1, bf, acc[1][j]);
      }
    }
    __syncthreads();
  }

  // Uniform per-tile destination select (scalar branch), then pure stores.
  const int half = (lane >> 4) << 3;
#pragma unroll
  for (int j = 0; j < 4; ++j) {
    const int colbase = tn + wn * 64 + j * 16;    // uniform
    const int which = colbase >> 10;              // 0=Q,1=K,2=V (uniform)
    const int dcolb = colbase & 1023;
    const int h = dcolb >> 6;                     // uniform
    const int s = (dcolb & 63) + (lane & 15);
    _Float16* dst = (which == 0) ? Q : (which == 1) ? K : V;
    const float scl = (which == 0) ? 0.125f : 1.0f;
    const float bv = bias[colbase + (lane & 15)];
#pragma unroll
    for (int i = 0; i < 2; ++i)
#pragma unroll
      for (int r = 0; r < 8; ++r) {
        const int grow = tm + wm * 32 + i * 16 + r + half;
        const int b = grow >> 11, n = grow & 2047;
        const size_t idx = (((size_t)(b * 16 + h)) * 2048 + n) * 64 + s;
        dst[idx] = (_Float16)((acc[i][j][r] + bv) * scl);
      }
  }
}

// ---------------------------------------------------------------------------
// Kernel 2: causal flash attention per (b,h). Block = 4 waves x 16 q-rows.
// K tile staged via TDM (tensor_load_to_lds) when available.
// ---------------------------------------------------------------------------
__global__ __launch_bounds__(128) void attn_kernel(
    const _Float16* __restrict__ Q, const _Float16* __restrict__ Kb,
    const _Float16* __restrict__ Vb, _Float16* __restrict__ Y) {
  __shared__ alignas(16) _Float16 Ks[64 * 72];
  __shared__ alignas(16) _Float16 Vt[64 * 72];
  __shared__ alignas(16) _Float16 Ps[4][16 * 72];

  const int tid = threadIdx.x, lane = tid & 31, wave = tid >> 5;
  const int bh = blockIdx.y;              // b*16 + h
  const int b = bh >> 4, h = bh & 15;
  const int q0 = blockIdx.x * 64;
  const int qrow = q0 + wave * 16;

  const size_t headBase = (size_t)bh * 2048 * 64;
  const _Float16* qh = Q + headBase;
  const _Float16* kh = Kb + headBase;
  const _Float16* vh = Vb + headBase;

  v16h qf0 = frag_a(qh + (size_t)qrow * 64 + 0,  64, lane);
  v16h qf1 = frag_a(qh + (size_t)qrow * 64 + 32, 64, lane);

  v8f O[4];
  float m[8], l[8];
#pragma unroll
  for (int c = 0; c < 4; ++c) O[c] = v8f{};
#pragma unroll
  for (int r = 0; r < 8; ++r) { m[r] = -3.0e38f; l[r] = 0.0f; }

  const int nkb = q0 / 64 + 1;
  const int lkey = tid >> 1;          // 0..63
  const int lc0  = (tid & 1) * 32;    // 0 or 32
  const int half = (lane >> 4) << 3;
#if USE_TDM
  const unsigned int ksAddr = (unsigned int)(uintptr_t)Ks;
#endif

  for (int kb = 0; kb < nkb; ++kb) {
    const int kbase = kb * 64;
    __syncthreads();
#if USE_TDM
    if (wave == 0)
      tdm_load_2d_f16(kh + (size_t)kbase * 64, ksAddr, 64, 64, 64);
#else
    {
      const _Float16* ksrc = kh + (size_t)(kbase + lkey) * 64 + lc0;
      _Float16* kdst = Ks + lkey * 72 + lc0;
#pragma unroll
      for (int q = 0; q < 4; ++q) *(v8h*)(kdst + q * 8) = *(const v8h*)(ksrc + q * 8);
    }
#endif
    {
      const _Float16* vsrc = vh + (size_t)(kbase + lkey) * 64 + lc0;
#pragma unroll
      for (int i = 0; i < 32; ++i) Vt[(lc0 + i) * 72 + lkey] = vsrc[i];
    }
#if USE_TDM
    if (wave == 0) __builtin_amdgcn_s_wait_tensorcnt(0);
#endif
    __syncthreads();

    // S = Q K^T  (16 q-rows x 64 keys, 4 col-tiles, K-dim 64 = 2 wmma steps)
    v8f S[4];
#pragma unroll
    for (int j = 0; j < 4; ++j) {
      v8f a = v8f{};
      a = WMMA_F32_F16(qf0, frag_b(Ks + (j * 16) * 72 + 0,  72, lane), a);
      a = WMMA_F32_F16(qf1, frag_b(Ks + (j * 16) * 72 + 32, 72, lane), a);
      S[j] = a;
    }

    // causal mask: col > row -> -inf
    const int rbase = qrow + half;
    const int cb = kbase + (lane & 15);
#pragma unroll
    for (int j = 0; j < 4; ++j)
#pragma unroll
      for (int r = 0; r < 8; ++r)
        if (cb + j * 16 > rbase + r) S[j][r] = -3.0e38f;

    // online softmax: row max/sum via 16-lane-group shfl_xor reductions
    float scale[8];
#pragma unroll
    for (int r = 0; r < 8; ++r) {
      float t = fmaxf(fmaxf(S[0][r], S[1][r]), fmaxf(S[2][r], S[3][r]));
      t = fmaxf(t, __shfl_xor(t, 1, 32));
      t = fmaxf(t, __shfl_xor(t, 2, 32));
      t = fmaxf(t, __shfl_xor(t, 4, 32));
      t = fmaxf(t, __shfl_xor(t, 8, 32));
      const float mn = fmaxf(m[r], t);
      scale[r] = __expf(m[r] - mn);
      m[r] = mn;
    }
#pragma unroll
    for (int j = 0; j < 4; ++j)
#pragma unroll
      for (int r = 0; r < 8; ++r) S[j][r] = __expf(S[j][r] - m[r]);
#pragma unroll
    for (int r = 0; r < 8; ++r) {
      float rs = S[0][r] + S[1][r] + S[2][r] + S[3][r];
      rs += __shfl_xor(rs, 1, 32);
      rs += __shfl_xor(rs, 2, 32);
      rs += __shfl_xor(rs, 4, 32);
      rs += __shfl_xor(rs, 8, 32);
      l[r] = l[r] * scale[r] + rs;
    }
#pragma unroll
    for (int c = 0; c < 4; ++c)
#pragma unroll
      for (int r = 0; r < 8; ++r) O[c][r] *= scale[r];

    // P: C-layout f32 -> per-wave LDS f16 -> A-layout fragments
    // (same-wave LDS ops are in-order per DScnt rules; no barrier needed)
    _Float16* pw = Ps[wave];
#pragma unroll
    for (int j = 0; j < 4; ++j)
#pragma unroll
      for (int r = 0; r < 8; ++r)
        pw[(r + half) * 72 + j * 16 + (lane & 15)] = (_Float16)S[j][r];
    v16h pf0 = frag_a(pw + 0,  72, lane);
    v16h pf1 = frag_a(pw + 32, 72, lane);

    // O += P V  (B operand from transposed V tile)
#pragma unroll
    for (int c = 0; c < 4; ++c) {
      O[c] = WMMA_F32_F16(pf0, frag_b(Vt + (c * 16) * 72 + 0,  72, lane), O[c]);
      O[c] = WMMA_F32_F16(pf1, frag_b(Vt + (c * 16) * 72 + 32, 72, lane), O[c]);
    }
  }

  // normalize and write y [B,N,D] f16, D index = h*64 + dim
#pragma unroll
  for (int r = 0; r < 8; ++r) l[r] = 1.0f / l[r];
#pragma unroll
  for (int c = 0; c < 4; ++c)
#pragma unroll
    for (int r = 0; r < 8; ++r) {
      const int nr = qrow + half + r;
      const int dim = c * 16 + (lane & 15);
      Y[((size_t)b * 2048 + nr) * 1024 + h * 64 + dim] = (_Float16)(O[c][r] * l[r]);
    }
}

// ---------------------------------------------------------------------------
// Kernel 3: out = y @ Wo^T + bo, f32 output. M=4096, N=1024, K=1024.
// A tile (f16 Y slice) staged via TDM when available.
// ---------------------------------------------------------------------------
__global__ __launch_bounds__(256) void proj_gemm(
    const _Float16* __restrict__ Yin, const float* __restrict__ W,
    const float* __restrict__ bias, float* __restrict__ out) {
  __shared__ alignas(16) _Float16 As[128 * 72];
  __shared__ alignas(16) _Float16 Bs[128 * 72];

  const int tid = threadIdx.x, lane = tid & 31, wave = tid >> 5;
  const int wm = wave & 3, wn = wave >> 2;
  const int tm = blockIdx.y * 128, tn = blockIdx.x * 128;

  v8f acc[2][4];
#pragma unroll
  for (int i = 0; i < 2; ++i)
#pragma unroll
    for (int j = 0; j < 4; ++j) acc[i][j] = v8f{};

  const int lrow = tid >> 1, lcol = (tid & 1) * 32;
  const float* wb = W + (size_t)(tn + lrow) * 1024 + lcol;
  _Float16* bsp = Bs + lrow * 72 + lcol;
#if USE_TDM
  const unsigned int asAddr = (unsigned int)(uintptr_t)As;
#else
  const _Float16* ya = Yin + (size_t)(tm + lrow) * 1024 + lcol;
  _Float16* asp = As + lrow * 72 + lcol;
#endif

  for (int k0 = 0; k0 < 1024; k0 += 64) {
#if USE_TDM
    if (wave == 0)
      tdm_load_2d_f16(Yin + (size_t)tm * 1024 + k0, asAddr, 64, 128, 1024);
#else
    v8h hy[4];
#pragma unroll
    for (int q = 0; q < 4; ++q) hy[q] = ((const v8h*)(ya + k0))[q];
#pragma unroll
    for (int q = 0; q < 4; ++q) *(v8h*)(asp + q * 8) = hy[q];
#endif
    float4 fb[8];
#pragma unroll
    for (int q = 0; q < 8; ++q) fb[q] = ((const float4*)(wb + k0))[q];
#pragma unroll
    for (int q = 0; q < 4; ++q)
      *(v8h*)(bsp + q * 8) = cvt8(fb[2 * q], fb[2 * q + 1]);
#if USE_TDM
    if (wave == 0) __builtin_amdgcn_s_wait_tensorcnt(0);
#endif
    __syncthreads();

#pragma unroll
    for (int t = 0; t < 2; ++t) {
      v16h af0 = frag_a(As + (wm * 32 + 0)  * 72 + t * 32, 72, lane);
      v16h af1 = frag_a(As + (wm * 32 + 16) * 72 + t * 32, 72, lane);
#pragma unroll
      for (int j = 0; j < 4; ++j) {
        v16h bf = frag_b(Bs + (wn * 64 + j * 16) * 72 + t * 32, 72, lane);
        acc[0][j] = WMMA_F32_F16(af0, bf, acc[0][j]);
        acc[1][j] = WMMA_F32_F16(af1, bf, acc[1][j]);
      }
    }
    __syncthreads();
  }

  const int half = (lane >> 4) << 3;
#pragma unroll
  for (int j = 0; j < 4; ++j) {
    const int gcol = tn + wn * 64 + j * 16 + (lane & 15);
    const float bv = bias[gcol];
#pragma unroll
    for (int i = 0; i < 2; ++i)
#pragma unroll
      for (int r = 0; r < 8; ++r) {
        const int grow = tm + wm * 32 + i * 16 + r + half;
        out[(size_t)grow * 1024 + gcol] = acc[i][j][r] + bv;
      }
  }
}

// ---------------------------------------------------------------------------
extern "C" void kernel_launch(void* const* d_in, const int* in_sizes, int n_in,
                              void* d_out, int out_size, void* d_ws, size_t ws_size,
                              hipStream_t stream) {
  (void)in_sizes; (void)n_in; (void)out_size; (void)ws_size;
  const float* x    = (const float*)d_in[0];
  // d_in[1] is the causal mask; causality is applied analytically in-kernel.
  const float* Wqkv = (const float*)d_in[2];
  const float* bqkv = (const float*)d_in[3];
  const float* Wo   = (const float*)d_in[4];
  const float* bo   = (const float*)d_in[5];
  float* out = (float*)d_out;

  const size_t HN = (size_t)2 * 16 * 2048 * 64;  // 4,194,304 elems per tensor
  _Float16* Q = (_Float16*)d_ws;
  _Float16* K = Q + HN;
  _Float16* V = K + HN;
  _Float16* Y = V + HN;

  qkv_gemm<<<dim3(24, 32), 256, 0, stream>>>(x, Wqkv, bqkv, Q, K, V);
  attn_kernel<<<dim3(32, 32), 128, 0, stream>>>(Q, K, V, Y);
  proj_gemm<<<dim3(8, 32), 256, 0, stream>>>(Y, Wo, bo, out);
}
